// MultiHeadAttention_5738076307818
// MI455X (gfx1250) — matrix operand order repaired and checked by reading, not compiled, not run on hardware
//
#include <hip/hip_runtime.h>
#include <cstdint>
#include <cstddef>

#define D_MODEL 2048
#define N_HEADS 16
#define D_K     128
#define BATCH   2
#define SEQ     2048

typedef __attribute__((ext_vector_type(8)))  __bf16 v8bf;
typedef __attribute__((ext_vector_type(16))) __bf16 v16bf;
typedef __attribute__((ext_vector_type(8)))  float  v8f;

#if __has_builtin(__builtin_amdgcn_global_load_async_to_lds_b128) && \
    __has_builtin(__builtin_amdgcn_s_wait_asynccnt)
#define HAVE_ASYNC_LDS 1
typedef int vi4 __attribute__((vector_size(16)));
typedef __attribute__((address_space(1))) vi4 gvi4;   // global int4
typedef __attribute__((address_space(3))) vi4 svi4;   // LDS int4
#endif

// Build a 16-element bf16 fragment from two 16-byte chunks.
__device__ __forceinline__ v16bf frag2(const __bf16* p0, const __bf16* p1) {
  v8bf lo = *(const v8bf*)p0;
  v8bf hi = *(const v8bf*)p1;
  v16bf r;
#pragma unroll
  for (int i = 0; i < 8; ++i) { r[i] = lo[i]; r[8 + i] = hi[i]; }
  return r;
}

__device__ __forceinline__ v8f wmma_bf16(v16bf a, v16bf b, v8f c) {
  return __builtin_amdgcn_wmma_f32_16x16x32_bf16(false, a, false, b, (short)0, c,
                                                 false, false);
}

// ---------------------------------------------------------------------------
// GEMM: C[M,N] = A[M,K] * W[N,K]^T  (einsum 'bld,ed->ble'), bf16 WMMA f32 acc.
// Double-buffered LDS (K-step 32); global loads for tile t+1 are issued
// before computing tile t so HBM latency overlaps WMMA.
// OUT_MODE 0: fp32 row-major [M, D_MODEL]
// OUT_MODE 1: bf16 head-major [B, H, L, D_K]
// ---------------------------------------------------------------------------
template <bool A_IS_F32, int OUT_MODE>
__global__ __launch_bounds__(256) void gemm_xwt(const void* __restrict__ Av,
                                                const float* __restrict__ W,
                                                void* __restrict__ Out, int K) {
  __shared__ __align__(16) __bf16 As[2][128 * 40];
  __shared__ __align__(16) __bf16 Bs[2][128 * 40];

  const int tid  = threadIdx.x;
  const int lane = tid & 31;
  const int wid  = tid >> 5;
  const int wm   = wid >> 1;   // 0..3 : 32-row slice
  const int wn   = wid & 1;    // 0..1 : 64-col slice
  const int hf   = lane >> 4;
  const int r    = lane & 15;

  const int m0 = blockIdx.y * 128;
  const int n0 = blockIdx.x * 128;

  // chunk maps (tile = 128 x 32)
  const int rA  = tid >> 3;          // fp32 float4 chunks: 8 per row
  const int cA4 = (tid & 7) * 4;
  const int rAb = tid >> 2;          // bf16 v8bf chunks: 4 per row
  const int cAb = (tid & 3) * 8;

  float4 aR[4], bR[4];
  v8bf   aRb[2];

  auto gload = [&](int kt) {
    if constexpr (A_IS_F32) {
      const float* A = (const float*)Av;
#pragma unroll
      for (int i = 0; i < 4; ++i)
        aR[i] = *(const float4*)(A + (size_t)(m0 + rA + 32 * i) * K + kt + cA4);
    } else {
      const __bf16* A = (const __bf16*)Av;
#pragma unroll
      for (int i = 0; i < 2; ++i)
        aRb[i] = *(const v8bf*)(A + (size_t)(m0 + rAb + 64 * i) * K + kt + cAb);
    }
#pragma unroll
    for (int i = 0; i < 4; ++i)
      bR[i] = *(const float4*)(W + (size_t)(n0 + rA + 32 * i) * K + kt + cA4);
  };

  auto stash = [&](int buf) {
    if constexpr (A_IS_F32) {
#pragma unroll
      for (int i = 0; i < 4; ++i) {
        __bf16* d = &As[buf][(rA + 32 * i) * 40 + cA4];
        d[0] = (__bf16)aR[i].x; d[1] = (__bf16)aR[i].y;
        d[2] = (__bf16)aR[i].z; d[3] = (__bf16)aR[i].w;
      }
    } else {
#pragma unroll
      for (int i = 0; i < 2; ++i)
        *(v8bf*)&As[buf][(rAb + 64 * i) * 40 + cAb] = aRb[i];
    }
#pragma unroll
    for (int i = 0; i < 4; ++i) {
      __bf16* d = &Bs[buf][(rA + 32 * i) * 40 + cA4];
      d[0] = (__bf16)bR[i].x; d[1] = (__bf16)bR[i].y;
      d[2] = (__bf16)bR[i].z; d[3] = (__bf16)bR[i].w;
    }
  };

  v8f acc[2][4];
#pragma unroll
  for (int a = 0; a < 2; ++a)
#pragma unroll
    for (int b = 0; b < 4; ++b)
#pragma unroll
      for (int i = 0; i < 8; ++i) acc[a][b][i] = 0.0f;

  gload(0);
  stash(0);
  __syncthreads();

  const int NT = K / 32;
  for (int t = 0; t < NT; ++t) {
    const int buf = t & 1;
    if (t + 1 < NT) gload((t + 1) * 32);   // overlap HBM with WMMA below

    v16bf af[2], bfc[4];
#pragma unroll
    for (int rt = 0; rt < 2; ++rt) {
      const __bf16* p = &As[buf][(wm * 32 + rt * 16 + r) * 40 + hf * 16];
      af[rt] = frag2(p, p + 8);
    }
#pragma unroll
    for (int ct = 0; ct < 4; ++ct) {
      const __bf16* p = &Bs[buf][(wn * 64 + ct * 16 + r) * 40 + hf * 16];
      bfc[ct] = frag2(p, p + 8);
    }
#pragma unroll
    for (int rt = 0; rt < 2; ++rt)
#pragma unroll
      for (int ct = 0; ct < 4; ++ct)
        acc[rt][ct] = wmma_bf16(af[rt], bfc[ct], acc[rt][ct]);

    if (t + 1 < NT) {
      stash(buf ^ 1);
      __syncthreads();
    }
  }

#pragma unroll
  for (int rt = 0; rt < 2; ++rt)
#pragma unroll
    for (int ct = 0; ct < 4; ++ct)
#pragma unroll
      for (int v = 0; v < 8; ++v) {
        int row = m0 + wm * 32 + rt * 16 + v + hf * 8;
        int col = n0 + wn * 64 + ct * 16 + r;
        float val = acc[rt][ct][v];
        if constexpr (OUT_MODE == 0) {
          ((float*)Out)[(size_t)row * D_MODEL + col] = val;
        } else {
          int bb = row >> 11, ll = row & (SEQ - 1);
          int hh = col >> 7,  dd = col & (D_K - 1);
          ((__bf16*)Out)[(((size_t)bb * N_HEADS + hh) * SEQ + ll) * D_K + dd] =
              (__bf16)val;
        }
      }
}

// ---------------------------------------------------------------------------
// RoPE in place on [B*H, SEQ, D_K] bf16. 64 threads per row.
// ---------------------------------------------------------------------------
__global__ __launch_bounds__(256) void rope_kernel(__bf16* X) {
  int row = blockIdx.x * 4 + (threadIdx.x >> 6);
  int d   = threadIdx.x & 63;
  int l   = row & (SEQ - 1);
  __bf16* p = X + (size_t)row * D_K;
  float x1 = (float)p[d];
  float x2 = (float)p[d + 64];
  float inv = __expf(-(float)d * (9.210340371976184f / 64.0f));  // 10000^(-d/64)
  float th  = (float)l * inv;
  float c = cosf(th), s = sinf(th);
  p[d]      = (__bf16)(x1 * c - x2 * s);
  p[d + 64] = (__bf16)(x1 * s + x2 * c);
}

// ---------------------------------------------------------------------------
// Flash attention (causal). Block = 128 Q rows of one (b,h); 8 waves x 16 rows.
// 64-key K/V tiles streamed through LDS; online softmax; bf16 WMMA.
// ---------------------------------------------------------------------------
__global__ __launch_bounds__(256) void flash_attn(const __bf16* __restrict__ Q,
                                                  const __bf16* __restrict__ Kb,
                                                  const __bf16* __restrict__ V,
                                                  __bf16* __restrict__ O) {
  __shared__ __align__(16) __bf16 Ks[64 * 136];   // [key][d]
  __shared__ __align__(16) __bf16 Vt[128 * 72];   // [d][key] (transposed)
  __shared__ __align__(16) __bf16 Ps[128 * 72];   // [q][key]

  const int tid  = threadIdx.x;
  const int lane = tid & 31;
  const int wid  = tid >> 5;
  const int hf   = lane >> 4;
  const int r    = lane & 15;
  const int bh   = blockIdx.y;
  const int b    = bh >> 4, h = bh & 15;
  const int qb   = blockIdx.x * 128;
  const size_t headOff = (size_t)bh * SEQ * D_K;

  // Preload this wave's 16 Q rows as A-fragments (4 k-steps over d=128).
  const __bf16* qrow = Q + headOff + (size_t)(qb + wid * 16 + r) * D_K;
  v16bf aQ[4];
#pragma unroll
  for (int kk = 0; kk < 4; ++kk) {
    const __bf16* p = qrow + kk * 32 + hf * 16;
    aQ[kk] = frag2(p, p + 8);
  }

  v8f oacc[8];
  float m_i[8], l_i[8];
#pragma unroll
  for (int dt = 0; dt < 8; ++dt)
#pragma unroll
    for (int i = 0; i < 8; ++i) oacc[dt][i] = 0.0f;
#pragma unroll
  for (int v = 0; v < 8; ++v) { m_i[v] = -1e30f; l_i[v] = 0.0f; }

  const float scale = 0.08838834764831845f;  // 1/sqrt(128)
  const int jmax = qb / 64 + 2;              // causal bound

  for (int j = 0; j < jmax; ++j) {
    const __bf16* Kp = Kb + headOff + (size_t)(j * 64) * D_K;
    const __bf16* Vp = V  + headOff + (size_t)(j * 64) * D_K;
#ifdef HAVE_ASYNC_LDS
#pragma unroll
    for (int i = 0; i < 4; ++i) {            // K tile via async DMA to LDS
      int c = tid + i * 256;
      int row = c >> 4, cc = (c & 15) * 8;
      __builtin_amdgcn_global_load_async_to_lds_b128(
          (gvi4*)(Kp + row * D_K + cc),
          (svi4*)&Ks[row * 136 + cc], 0, 0);
    }
#pragma unroll
    for (int i = 0; i < 4; ++i) {            // V tile: transpose via VGPRs
      int c = tid + i * 256;
      int row = c >> 4, cc = (c & 15) * 8;
      v8bf vv = *(const v8bf*)(Vp + row * D_K + cc);
#pragma unroll
      for (int e = 0; e < 8; ++e) Vt[(cc + e) * 72 + row] = vv[e];
    }
    __builtin_amdgcn_s_wait_asynccnt(0);
#else
#pragma unroll
    for (int i = 0; i < 4; ++i) {            // 1024 chunks of 8 bf16
      int c = tid + i * 256;
      int row = c >> 4, cc = (c & 15) * 8;
      *(v8bf*)&Ks[row * 136 + cc] = *(const v8bf*)(Kp + row * D_K + cc);
      v8bf vv = *(const v8bf*)(Vp + row * D_K + cc);
#pragma unroll
      for (int e = 0; e < 8; ++e) Vt[(cc + e) * 72 + row] = vv[e];
    }
#endif
    __syncthreads();

    // S = Q K^T, software-pipelined B-fragment loads
    v8f sacc[4];
#pragma unroll
    for (int ct = 0; ct < 4; ++ct)
#pragma unroll
      for (int i = 0; i < 8; ++i) sacc[ct][i] = 0.0f;

    v16bf bk[4], bkn[4];
#pragma unroll
    for (int ct = 0; ct < 4; ++ct) {
      const __bf16* p = &Ks[(ct * 16 + r) * 136 + hf * 16];
      bk[ct] = frag2(p, p + 8);
    }
#pragma unroll
    for (int kk = 0; kk < 4; ++kk) {
      if (kk < 3) {
#pragma unroll
        for (int ct = 0; ct < 4; ++ct) {
          const __bf16* p = &Ks[(ct * 16 + r) * 136 + (kk + 1) * 32 + hf * 16];
          bkn[ct] = frag2(p, p + 8);
        }
      }
#pragma unroll
      for (int ct = 0; ct < 4; ++ct)
        sacc[ct] = wmma_bf16(aQ[kk], bk[ct], sacc[ct]);
#pragma unroll
      for (int ct = 0; ct < 4; ++ct) bk[ct] = bkn[ct];
    }

    // scale + causal mask
#pragma unroll
    for (int ct = 0; ct < 4; ++ct) {
      int col = j * 64 + ct * 16 + r;
#pragma unroll
      for (int v = 0; v < 8; ++v) {
        int row = qb + wid * 16 + v + hf * 8;
        float s = sacc[ct][v] * scale;
        sacc[ct][v] = (col > row) ? -1e30f : s;
      }
    }

    // online softmax (row reductions across the 16-lane half)
    float sc[8];
#pragma unroll
    for (int v = 0; v < 8; ++v) {
      float mx = fmaxf(fmaxf(sacc[0][v], sacc[1][v]),
                       fmaxf(sacc[2][v], sacc[3][v]));
      mx = fmaxf(mx, __shfl_xor(mx, 1, 32));
      mx = fmaxf(mx, __shfl_xor(mx, 2, 32));
      mx = fmaxf(mx, __shfl_xor(mx, 4, 32));
      mx = fmaxf(mx, __shfl_xor(mx, 8, 32));
      float mnew = fmaxf(m_i[v], mx);
      sc[v] = __expf(m_i[v] - mnew);
      m_i[v] = mnew;
      float rs = 0.0f;
#pragma unroll
      for (int ct = 0; ct < 4; ++ct) {
        float pv = __expf(sacc[ct][v] - mnew);
        sacc[ct][v] = pv;
        rs += pv;
      }
      rs += __shfl_xor(rs, 1, 32);
      rs += __shfl_xor(rs, 2, 32);
      rs += __shfl_xor(rs, 4, 32);
      rs += __shfl_xor(rs, 8, 32);
      l_i[v] = l_i[v] * sc[v] + rs;
    }

    // store P (bf16) and rescale O accumulators
#pragma unroll
    for (int ct = 0; ct < 4; ++ct)
#pragma unroll
      for (int v = 0; v < 8; ++v)
        Ps[(wid * 16 + v + hf * 8) * 72 + ct * 16 + r] = (__bf16)sacc[ct][v];
#pragma unroll
    for (int dt = 0; dt < 8; ++dt)
#pragma unroll
      for (int v = 0; v < 8; ++v) oacc[dt][v] *= sc[v];
    __syncthreads();

    // O += P * V, pipelined V-fragment loads
#pragma unroll
    for (int kk = 0; kk < 2; ++kk) {
      const __bf16* pp = &Ps[(wid * 16 + r) * 72 + kk * 32 + hf * 16];
      v16bf ap = frag2(pp, pp + 8);
      const __bf16* v0 = &Vt[r * 72 + kk * 32 + hf * 16];
      v16bf bv = frag2(v0, v0 + 8);
#pragma unroll
      for (int dt = 0; dt < 8; ++dt) {
        v16bf nxt;
        if (dt < 7) {
          const __bf16* vp = &Vt[((dt + 1) * 16 + r) * 72 + kk * 32 + hf * 16];
          nxt = frag2(vp, vp + 8);
        }
        oacc[dt] = wmma_bf16(ap, bv, oacc[dt]);
        bv = nxt;
      }
    }
    __syncthreads();
  }

  // normalize + write to [B, L, E] bf16 for the output projection
#pragma unroll
  for (int dt = 0; dt < 8; ++dt)
#pragma unroll
    for (int v = 0; v < 8; ++v) {
      int l = qb + wid * 16 + v + hf * 8;
      int e = h * D_K + dt * 16 + r;
      O[((size_t)b * SEQ + l) * D_MODEL + e] = (__bf16)(oacc[dt][v] / l_i[v]);
    }
}

// ---------------------------------------------------------------------------
extern "C" void kernel_launch(void* const* d_in, const int* in_sizes, int n_in,
                              void* d_out, int out_size, void* d_ws,
                              size_t ws_size, hipStream_t stream) {
  (void)in_sizes; (void)n_in; (void)out_size; (void)ws_size;
  const float* query = (const float*)d_in[0];
  const float* key   = (const float*)d_in[1];
  const float* value = (const float*)d_in[2];
  // d_in[3] = mask (causal; handled analytically)
  const float* w_q = (const float*)d_in[4];
  const float* w_k = (const float*)d_in[5];
  const float* w_v = (const float*)d_in[6];
  const float* w_o = (const float*)d_in[7];
  float* out = (float*)d_out;

  const size_t headElems = (size_t)BATCH * N_HEADS * SEQ * D_K;  // 8Mi
  __bf16* Qbf = (__bf16*)d_ws;
  __bf16* Kbf = Qbf + headElems;
  __bf16* Vbf = Kbf + headElems;
  __bf16* Obf = Vbf + headElems;

  dim3 g(D_MODEL / 128, (BATCH * SEQ) / 128);  // (16, 32)
  gemm_xwt<true, 1><<<g, 256, 0, stream>>>(query, w_q, Qbf, D_MODEL);
  gemm_xwt<true, 1><<<g, 256, 0, stream>>>(key,   w_k, Kbf, D_MODEL);
  gemm_xwt<true, 1><<<g, 256, 0, stream>>>(value, w_v, Vbf, D_MODEL);

  rope_kernel<<<(BATCH * N_HEADS * SEQ) / 4, 256, 0, stream>>>(Qbf);
  rope_kernel<<<(BATCH * N_HEADS * SEQ) / 4, 256, 0, stream>>>(Kbf);

  flash_attn<<<dim3(SEQ / 128, BATCH * N_HEADS), 256, 0, stream>>>(Qbf, Kbf,
                                                                   Vbf, Obf);

  gemm_xwt<false, 0><<<g, 256, 0, stream>>>(Obf, w_o, out, D_MODEL);
}
